// StackedRNNNet_50165218017845
// MI455X (gfx1250) — compile-verified, matrix operand order checked by reading, not run on hardware
//
#include <hip/hip_runtime.h>
#include <hip/hip_bf16.h>

// ---------------------------------------------------------------------------
// Stacked LSTM (3 layers, B=128, T=256, D=512, H=1024) for gfx1250 (MI455X).
//
//  * prep: f32->bf16 convert of x; pack [W;U] -> bf16 [Din+H, 4H] with
//    columns unit-major (col = unit*4 + gate, gates i,f,g,o); pack bias.
//  * per (layer,timestep): gates = [x_t|h] @ [W;U] + b via
//    v_wmma_f32_16x16x32_bf16, fused sigmoid/tanh cell update.
//    256 WGs (16 cols each) x 8 waves (16 rows each). K unrolled by 64:
//    2 WMMAs per loop trip, double-buffered LDS B staging -> single
//    __syncthreads per trip, global_prefetch_b8 two tiles ahead.
//    Weights (44 MB bf16) stay resident in the 192 MB L2 across all
//    768 sequential timesteps; HBM traffic is only activations (~0.5 GB).
// ---------------------------------------------------------------------------

#define B_ 128
#define T_ 256
#define D_ 512
#define H_ 1024
#define NC 4096   // 4*H packed gate columns

typedef __attribute__((ext_vector_type(16))) __bf16        v16bf;
typedef __attribute__((ext_vector_type(8)))  float         v8f;
typedef __attribute__((ext_vector_type(8)))  unsigned int  vu8;

// ------------------------------- prep kernels ------------------------------

__global__ __launch_bounds__(256) void f32_to_bf16_kernel(
    const float* __restrict__ src, __hip_bfloat16* __restrict__ dst, size_t n) {
  size_t i = (size_t)blockIdx.x * blockDim.x + threadIdx.x;
  if (i < n) dst[i] = __float2bfloat16(src[i]);
}

// WU[k][u*4+g] = (k < Din) ? W_g[k][u] : U_g[k-Din][u]   (bf16)
__global__ __launch_bounds__(256) void pack_wu_kernel(
    const float* __restrict__ Wi, const float* __restrict__ Ui,
    const float* __restrict__ Wf, const float* __restrict__ Uf,
    const float* __restrict__ Wg, const float* __restrict__ Ug,
    const float* __restrict__ Wo, const float* __restrict__ Uo,
    __hip_bfloat16* __restrict__ WU, int Din, int Ktot) {
  size_t idx = (size_t)blockIdx.x * blockDim.x + threadIdx.x;
  size_t total = (size_t)Ktot * NC;
  if (idx >= total) return;
  int n = (int)(idx & (NC - 1));
  int k = (int)(idx >> 12);           // idx / 4096
  int u = n >> 2;
  int g = n & 3;
  const float* Wsel = (g == 0) ? Wi : (g == 1) ? Wf : (g == 2) ? Wg : Wo;
  const float* Usel = (g == 0) ? Ui : (g == 1) ? Uf : (g == 2) ? Ug : Uo;
  float v = (k < Din) ? Wsel[(size_t)k * H_ + u]
                      : Usel[(size_t)(k - Din) * H_ + u];
  WU[idx] = __float2bfloat16(v);
}

__global__ __launch_bounds__(256) void pack_bias_kernel(
    const float* __restrict__ bi, const float* __restrict__ bf_,
    const float* __restrict__ bg, const float* __restrict__ bo,
    float* __restrict__ bp) {
  int n = blockIdx.x * blockDim.x + threadIdx.x;
  if (n >= NC) return;
  int u = n >> 2, g = n & 3;
  const float* bsel = (g == 0) ? bi : (g == 1) ? bf_ : (g == 2) ? bg : bo;
  bp[n] = bsel[u];
}

__global__ __launch_bounds__(256) void init_state_kernel(
    __hip_bfloat16* __restrict__ h, float* __restrict__ c, int n) {
  int i = blockIdx.x * blockDim.x + threadIdx.x;
  if (i >= n) return;
  h[i] = __float2bfloat16(0.0f);
  c[i] = 0.0f;
}

// ------------------------------- step kernel -------------------------------
// One timestep of one layer:
//   gates[B, 4H] = [x_t | h_prev] @ WU + bias ; then cell/hidden update.
// blockIdx.x in [0,256): owns output columns n0 = blockIdx.x*16.
// wave w in [0,8): owns output rows m0 = w*16.  K = Din + H, 64 per trip.
__global__ __launch_bounds__(256) void lstm_step_kernel(
    const __hip_bfloat16* __restrict__ xin,   // [B, T, Din] layer input seq
    const __hip_bfloat16* __restrict__ hin,   // [B, H] previous hidden
    const __hip_bfloat16* __restrict__ WU,    // [Ktot, 4096] packed weights
    const float*          __restrict__ biasp, // [4096] packed bias
    float*                __restrict__ cstate,// [B, H] cell state (in place)
    __hip_bfloat16*       __restrict__ hout,  // [B, H] new hidden (ping-pong)
    __hip_bfloat16*       __restrict__ hseq,  // [B, T, H] layer output (or null)
    float*                __restrict__ out_xs,// [B, T, H] f32 (last layer only)
    float*                __restrict__ out_cs,// [T, B, H] f32 (last layer only)
    int t, int Din, int Ktot) {
  // Double-buffered B staging in WMMA fragment order:
  //   ldsB[buf][ksub(32-wide)][lane][element]
  __shared__ __align__(16) __hip_bfloat16 ldsB[2][2][32][16];
  __shared__ float ldsAcc[8][16][16];   // per-wave 16x16 f32 gate tile

  const int tid    = threadIdx.x;
  const int w      = tid >> 5;        // wave id 0..7
  const int l      = tid & 31;        // lane id
  const int half   = l >> 4;          // 0: lanes 0-15, 1: lanes 16-31
  const int lane16 = l & 15;
  const int m0     = w << 4;          // batch-row tile base
  const int n0     = blockIdx.x << 4; // packed-gate column tile base
  const int m      = m0 + lane16;     // this lane's A-matrix row (batch)

  // Hoisted A row base pointers (x part and h part).
  const __hip_bfloat16* const xrow = xin + ((size_t)m * T_ + t) * Din;
  const __hip_bfloat16* const hrow = hin + (size_t)m * H_;

  // Stage one 64x16 B tile (rows kb..kb+63, cols n0..n0+15) into buffer buf.
  // 1024 bf16 = 512 u32 pairs; 256 threads x 2 pairs. Fragment mapping for
  // 16-bit B (wave32): row Kl of a 32-wide subtile lands at
  //   element e = (Kl&7) | ((Kl>>4)<<3),  lane = ((Kl>>3)&1)*16 + n.
  auto stageB = [&](int buf, int kb) {
#pragma unroll
    for (int it = 0; it < 2; ++it) {
      int p  = tid + (it << 8);       // 0..511
      int K  = p >> 3;                // 0..63
      int n  = (p & 7) << 1;          // 0,2,...,14
      int s  = K >> 5;                // 32-wide subtile
      int Kl = K & 31;
      int e     = (Kl & 7) | ((Kl >> 4) << 3);
      int lane0 = (((Kl >> 3) & 1) << 4) | n;
      unsigned v = *(const unsigned*)(&WU[(size_t)(kb + K) * NC + n0 + n]);
      ldsB[buf][s][lane0][e]     = ((const __hip_bfloat16*)&v)[0];
      ldsB[buf][s][lane0 + 1][e] = ((const __hip_bfloat16*)&v)[1];
    }
  };

  // A fragment for a 32-wide K chunk at kb. 16-bit A 16x32 layout:
  //   lanes 0-15 : elems 0-7 -> K 0-7,  elems 8-15 -> K 16-23
  //   lanes 16-31: elems 0-7 -> K 8-15, elems 8-15 -> K 24-31
  auto loadA = [&](int kb) -> v16bf {
    const __hip_bfloat16* ab = (kb < Din) ? (xrow + kb) : (hrow + (kb - Din));
    const uint4 a0 = *(const uint4*)(ab + (half << 3));
    const uint4 a1 = *(const uint4*)(ab + 16 + (half << 3));
    vu8 r;
    r[0] = a0.x; r[1] = a0.y; r[2] = a0.z; r[3] = a0.w;
    r[4] = a1.x; r[5] = a1.y; r[6] = a1.z; r[7] = a1.w;
    return __builtin_bit_cast(v16bf, r);
  };

  auto loadB = [&](int buf, int s) -> v16bf {
    const uint4* p = (const uint4*)&ldsB[buf][s][l][0];
    const uint4 b0 = p[0], b1 = p[1];
    vu8 r;
    r[0] = b0.x; r[1] = b0.y; r[2] = b0.z; r[3] = b0.w;
    r[4] = b1.x; r[5] = b1.y; r[6] = b1.z; r[7] = b1.w;
    return __builtin_bit_cast(v16bf, r);
  };

  v8f acc = {};
  const int kiters = Ktot >> 6;       // 64 K-values per trip

  stageB(0, 0);
  __syncthreads();

  for (int ki = 0; ki < kiters; ++ki) {
    const int kb  = ki << 6;
    const int cur = ki & 1;

    // Stage next tile into the other buffer; prefetch two tiles ahead.
    if (ki + 1 < kiters) {
      stageB(cur ^ 1, kb + 64);
      if (ki + 2 < kiters) {
        int Kp = tid >> 2;            // 0..63: one line per next-next row
        __builtin_prefetch(&WU[(size_t)(kb + 128 + Kp) * NC + n0], 0, 1);
      }
    }

    v16bf a0 = loadA(kb);
    v16bf b0 = loadB(cur, 0);
    acc = __builtin_amdgcn_wmma_f32_16x16x32_bf16(
        false, a0, false, b0, (short)0, acc, false, false);

    v16bf a1 = loadA(kb + 32);
    v16bf b1 = loadB(cur, 1);
    acc = __builtin_amdgcn_wmma_f32_16x16x32_bf16(
        false, a1, false, b1, (short)0, acc, false, false);

    // Single barrier: publishes next buffer's stores AND retires this
    // buffer's reads before it is overwritten in the next trip.
    __syncthreads();
  }

  // ---- bias add + scatter accumulator tile to LDS for the gate fusion.
  // C/D layout: VGPR r, lanes 0-15 -> M=r, lanes 16-31 -> M=r+8; N = lane%16.
  const float bn = biasp[n0 + lane16];
#pragma unroll
  for (int r = 0; r < 8; ++r)
    ldsAcc[w][r + (half << 3)][lane16] = acc[r] + bn;
  __syncthreads();

  // ---- fused LSTM cell update. Tile = 16 batch rows x 4 units x 4 gates.
#pragma unroll
  for (int j = 0; j < 2; ++j) {
    int q   = l + (j << 5);
    int row = q >> 2;
    int ul  = q & 3;
    float gi = ldsAcc[w][row][(ul << 2) + 0];
    float gf = ldsAcc[w][row][(ul << 2) + 1];
    float gg = ldsAcc[w][row][(ul << 2) + 2];
    float go = ldsAcc[w][row][(ul << 2) + 3];
    float ig = 1.0f / (1.0f + __expf(-gi));
    float fg = 1.0f / (1.0f + __expf(-gf));
    float gt = tanhf(gg);
    float og = 1.0f / (1.0f + __expf(-go));
    int bi_ = m0 + row;                  // batch index
    int u_  = (n0 >> 2) + ul;            // hidden unit index
    size_t cidx = (size_t)bi_ * H_ + u_;
    float cn = fg * cstate[cidx] + ig * gt;
    float hn = og * tanhf(cn);
    cstate[cidx] = cn;
    hout[cidx]   = __float2bfloat16(hn);
    if (hseq) hseq[((size_t)bi_ * T_ + t) * H_ + u_] = __float2bfloat16(hn);
    if (out_xs) {
      out_xs[((size_t)bi_ * T_ + t) * H_ + u_] = hn;   // xs [B,T,H]
      out_cs[((size_t)t * B_ + bi_) * H_ + u_] = cn;   // cs [T,B,H]
    }
  }
}

// --------------------------------- launcher --------------------------------

extern "C" void kernel_launch(void* const* d_in, const int* in_sizes, int n_in,
                              void* d_out, int out_size, void* d_ws, size_t ws_size,
                              hipStream_t stream) {
  (void)in_sizes; (void)n_in; (void)out_size; (void)ws_size;
  const float* x = (const float*)d_in[0];

  // ---- carve workspace (all regions fully (re)written every call)
  char* ws = (char*)d_ws;
  size_t off = 0;
  auto carve = [&](size_t bytes) -> char* {
    char* p = ws + off;
    off = (off + bytes + 255) & ~(size_t)255;
    return p;
  };
  __hip_bfloat16* xbf   = (__hip_bfloat16*)carve((size_t)B_ * T_ * D_ * 2);
  __hip_bfloat16* hseq0 = (__hip_bfloat16*)carve((size_t)B_ * T_ * H_ * 2);
  __hip_bfloat16* hseq1 = (__hip_bfloat16*)carve((size_t)B_ * T_ * H_ * 2);
  int Din[3] = {D_, H_, H_};
  __hip_bfloat16* WU[3];
  float* bp[3];
  for (int l = 0; l < 3; ++l) {
    WU[l] = (__hip_bfloat16*)carve((size_t)(Din[l] + H_) * NC * 2);
    bp[l] = (float*)carve((size_t)NC * 4);
  }
  __hip_bfloat16* hA = (__hip_bfloat16*)carve((size_t)B_ * H_ * 2);
  __hip_bfloat16* hB = (__hip_bfloat16*)carve((size_t)B_ * H_ * 2);
  float*          cs = (float*)carve((size_t)B_ * H_ * 4);

  // ---- prep: x -> bf16
  {
    size_t n = (size_t)B_ * T_ * D_;
    f32_to_bf16_kernel<<<(unsigned)((n + 255) / 256), 256, 0, stream>>>(x, xbf, n);
  }
  // ---- prep: pack weights + bias per layer (gate order in d_in: i,f,g,o;
  //      per gate: W, U, b  =>  layer l base index = 1 + l*12 + g*3)
  for (int l = 0; l < 3; ++l) {
    const float *Wg[4], *Ug[4], *bg[4];
    for (int g = 0; g < 4; ++g) {
      int base = 1 + l * 12 + g * 3;
      Wg[g] = (const float*)d_in[base];
      Ug[g] = (const float*)d_in[base + 1];
      bg[g] = (const float*)d_in[base + 2];
    }
    int Ktot = Din[l] + H_;
    size_t tot = (size_t)Ktot * NC;
    pack_wu_kernel<<<(unsigned)((tot + 255) / 256), 256, 0, stream>>>(
        Wg[0], Ug[0], Wg[1], Ug[1], Wg[2], Ug[2], Wg[3], Ug[3],
        WU[l], Din[l], Ktot);
    pack_bias_kernel<<<NC / 256, 256, 0, stream>>>(bg[0], bg[1], bg[2], bg[3], bp[l]);
  }

  float* out_xs = (float*)d_out;                       // [B,T,H]
  float* out_cs = out_xs + (size_t)B_ * T_ * H_;       // [T,B,H]

  const __hip_bfloat16* layer_in[3]  = {xbf, hseq0, hseq1};
  __hip_bfloat16*       layer_out[3] = {hseq0, hseq1, nullptr};

  for (int l = 0; l < 3; ++l) {
    init_state_kernel<<<(B_ * H_ + 255) / 256, 256, 0, stream>>>(hA, cs, B_ * H_);
    int Ktot = Din[l] + H_;
    bool last = (l == 2);
    for (int t = 0; t < T_; ++t) {
      const __hip_bfloat16* hin = (t & 1) ? hB : hA;   // ping-pong hidden
      __hip_bfloat16*       ho  = (t & 1) ? hA : hB;
      lstm_step_kernel<<<NC / 16, 256, 0, stream>>>(
          layer_in[l], hin, WU[l], bp[l], cs, ho, layer_out[l],
          last ? out_xs : nullptr, last ? out_cs : nullptr,
          t, Din[l], Ktot);
    }
  }
}
